// GaussianEmbedding_36163624632345
// MI455X (gfx1250) — matrix operand order repaired and checked by reading, not compiled, not run on hardware
//
#include <hip/hip_runtime.h>
#include <hip/hip_bf16.h>
#include <stdint.h>

// Problem dims (fixed by reference setup_inputs)
#define BB     32
#define TTEXT  512
#define NTOK   514      // TTEXT + 2 pad
#define NP     544      // padded to 17 * 32
#define NCHUNK 17
#define TT     4096
#define VV     100
#define DD     512

#define EPSV        1e-6f
#define LOG_SQRT2PI 0.9189385332046727f
#define NEG_BIG     (-1.0e30f)
#define BIG_END     3.4e38f

#define EROW   (NP / 2)      // 272 uints (half2 pairs) per gathered-embedding column
#define LROW   20            // padded LDS row stride (uints): 80B rows -> 16B-aligned
                             // b128 quarters AND conflict-free (20d mod 64 distinct)
#define LBUF   (256 * LROW)  // one LDS E-buffer (uints)

typedef __attribute__((ext_vector_type(16))) _Float16 v16h;
typedef __attribute__((ext_vector_type(8)))  float    v8f;

// ---------------- Kernel 1: per-batch token params ----------------
__global__ __launch_bounds__(256)
void ge_params(const int* __restrict__ text, const int* __restrict__ durs,
               float* __restrict__ gc, float* __restrict__ ginv,
               float* __restrict__ glse, float* __restrict__ gends,
               int* __restrict__ gtok)
{
    __shared__ float endsS[NTOK];
    const int b = blockIdx.x;
    const int tid = threadIdx.x;

    if (tid == 0) {
        float run = 0.f;
        for (int n = 0; n < NTOK; ++n) {
            float dv = (n < TTEXT) ? (float)durs[b * TTEXT + n] : 0.f;
            run += dv;
            endsS[n] = run;
        }
    }
    __syncthreads();

    for (int n = tid; n < NP; n += 256) {
        if (n < NTOK) {
            float dv   = (n < TTEXT) ? (float)durs[b * TTEXT + n] : 0.f;
            float eprv = (n > 0) ? endsS[n - 1] : 0.f;
            float c    = dv * 0.5f + eprv;          // durs/2 + prior cumsum
            float sig  = dv * 0.5f + EPSV;          // durs/SIGMA_C + EPS, SIGMA_C=2
            int   tok  = (n < TTEXT) ? text[b * TTEXT + n] : 0;
            float lse  = -__logf(sig) - LOG_SQRT2PI;
            if (tok == 0) lse = NEG_BIG;            // PAD token -> zero weight
            gc  [b * NP + n] = c;
            ginv[b * NP + n] = 1.f / sig;
            glse[b * NP + n] = lse;
            gends[b * NP + n] = endsS[n];
            gtok[b * NP + n] = tok;
        } else {
            gc  [b * NP + n] = 0.f;
            ginv[b * NP + n] = 0.f;
            glse[b * NP + n] = NEG_BIG;             // tail -> zero weight
            gends[b * NP + n] = BIG_END;            // never counted by search
            gtok[b * NP + n] = 0;
        }
    }
}

// ------ Kernel 2: gathered embedding, f16, K-pair-major transpose ------
// EbufT[b][d][np] = half2( E(2np, d), E(2np+1, d) ), E(n,d)=embed_w[tok[b][n]][d]
__global__ __launch_bounds__(256)
void ge_embT(const float* __restrict__ embed_w, const int* __restrict__ gtok,
             uint32_t* __restrict__ ebufT)
{
    const int i = blockIdx.x * 256 + threadIdx.x;
    const int total = BB * DD * EROW;
    if (i >= total) return;
    const int b  = i / (DD * EROW);
    const int r  = i % (DD * EROW);
    const int d  = r / EROW;
    const int np = r % EROW;
    const int t0 = gtok[b * NP + 2 * np];
    const int t1 = gtok[b * NP + 2 * np + 1];
    const float f0 = embed_w[t0 * DD + d];
    const float f1 = embed_w[t1 * DD + d];
    union { uint32_t u; _Float16 h[2]; } cv;
    cv.h[0] = (_Float16)f0;
    cv.h[1] = (_Float16)f1;
    ebufT[i] = cv.u;
}

// ------------- Kernel 3: per-frame normalizer + pad-frame flag -------------
__global__ __launch_bounds__(128)
void ge_rows(const float* __restrict__ gc, const float* __restrict__ ginv,
             const float* __restrict__ glse, const float* __restrict__ gends,
             const int* __restrict__ gtok,
             float* __restrict__ ginvn, float* __restrict__ gfpad)
{
    __shared__ float sc[NP], sv[NP], sl[NP], se[NP];
    const int b = blockIdx.y;
    const int t = blockIdx.x * 128 + threadIdx.x;

    for (int i = threadIdx.x; i < NP; i += 128) {
        sc[i] = gc  [b * NP + i];
        sv[i] = ginv[b * NP + i];
        sl[i] = glse[b * NP + i];
        se[i] = gends[b * NP + i];
    }
    __syncthreads();

    const float tf = (float)t + 0.5f;
    float s = 0.f;
    #pragma unroll 4
    for (int n = 0; n < NP; ++n) {
        float z = (tf - sc[n]) * sv[n];
        s += __expf(fmaf(-0.5f * z, z, sl[n]));
    }
    // searchsorted(ends, t, side='right') == count(ends <= t)
    const float f = (float)t;
    int lo = 0, hi = NP;
    while (lo < hi) {
        int mid = (lo + hi) >> 1;
        if (se[mid] <= f) lo = mid + 1; else hi = mid;
    }
    int idx = lo < (NTOK - 1) ? lo : (NTOK - 1);
    ginvn[b * TT + t] = 1.f / (s + EPSV);
    gfpad[b * TT + t] = (gtok[b * NP + idx] == 0) ? 1.f : 0.f;
}

// ---------------- Kernel 4: fused Gaussian-weight WMMA GEMM ----------------
// Block: 256 threads = 8 waves, tile 128 T x 256 D.
// Waves: wt = wave&3 (32 T-rows each), wd = wave>>2 (128 D-cols each).
// Per wave: 2 A fragments, 8 B column groups (double-buffered, ds_load_b128),
// 16 accumulators, 16 WMMA/chunk. E chunks staged to double-buffered LDS with
// global_load_async_to_lds_b128 overlapped with compute (ASYNCcnt + barrier).
__global__ __launch_bounds__(256)
void ge_gemm(const float* __restrict__ gc, const float* __restrict__ ginv,
             const float* __restrict__ glse,
             const float* __restrict__ ginvn, const float* __restrict__ gfpad,
             const uint32_t* __restrict__ ebufT,
             float* __restrict__ out)
{
    __shared__ float sc[NP], sv[NP], sl[NP];
    __shared__ uint32_t elds[2 * LBUF];   // [buf][d][kpair], stride-20 rows

    const int b   = blockIdx.z;
    const int t0  = blockIdx.y * 128;
    const int d0  = blockIdx.x * 256;
    const int tid = threadIdx.x;
    const int lane = tid & 31;
    const int wave = tid >> 5;
    const int wt = wave & 3;          // T sub-tile (32 rows)
    const int wd = wave >> 2;         // D sub-tile (128 cols)
    const bool laneHi = lane >= 16;

    for (int i = tid; i < NP; i += 256) {
        sc[i] = gc  [b * NP + i];
        sv[i] = ginv[b * NP + i];
        sl[i] = glse[b * NP + i];
    }

    // per-lane frame (A-matrix row) params for the wave's two 16-row groups
    const int mA = lane & 15;
    float tfv[2], inrm[2];
    bool  fpv[2];
    #pragma unroll
    for (int ar = 0; ar < 2; ++ar) {
        const int tA = t0 + wt * 32 + ar * 16 + mA;
        tfv[ar]  = (float)tA + 0.5f;
        inrm[ar] = ginvn[b * TT + tA];
        fpv[ar]  = gfpad[b * TT + tA] > 0.5f;
    }

    v8f acc[2][8];
    #pragma unroll
    for (int ar = 0; ar < 2; ++ar)
        #pragma unroll
        for (int j = 0; j < 8; ++j)
            acc[ar][j] = (v8f){0.f,0.f,0.f,0.f,0.f,0.f,0.f,0.f};

    // async-staging addresses: this thread owns embedding column d0+tid
    const uint64_t ebase   = (uint64_t)(uintptr_t)ebufT;
    const uint32_t erowOff = (uint32_t)((b * DD + d0 + tid) * EROW) * 4u; // bytes

    // issue chunk 0 into buffer 0
    {
        const uint32_t la0 = (uint32_t)(uintptr_t)&elds[tid * LROW];
        #pragma unroll
        for (int q = 0; q < 4; ++q) {
            uint32_t la = la0 + q * 16u;
            uint32_t va = erowOff + q * 16u;
            asm volatile("global_load_async_to_lds_b128 %0, %1, %2"
                         :: "v"(la), "v"(va), "s"(ebase) : "memory");
        }
    }

    const int nl = lane & 15;

    for (int ic = 0; ic < NCHUNK; ++ic) {
        const int cur = ic & 1;
        // own async writes for chunk ic complete
        asm volatile("s_wait_asynccnt 0" ::: "memory");
        // all waves' writes complete; all waves done reading the other buffer
        __syncthreads();

        // overlap: issue chunk ic+1 into the other buffer
        if (ic + 1 < NCHUNK) {
            const uint32_t la0 = (uint32_t)(uintptr_t)&elds[(1 - cur) * LBUF + tid * LROW];
            const uint32_t va0 = erowOff + (uint32_t)((ic + 1) * 16 * 4);
            #pragma unroll
            for (int q = 0; q < 4; ++q) {
                uint32_t la = la0 + q * 16u;
                uint32_t va = va0 + q * 16u;
                asm volatile("global_load_async_to_lds_b128 %0, %1, %2"
                             :: "v"(la), "v"(va), "s"(ebase) : "memory");
            }
        }

        const uint32_t* ebuf = &elds[cur * LBUF];

        // B-fragment loader: lane&15 = N column; lanes<16 hold K 0..15
        // (uints 0..7 of the row), lanes>=16 hold K 16..31 (uints 8..15).
        // Row base dIdx*80B and +32B offset are 16B-aligned -> 2x ds_load_b128.
        auto loadBM = [&](int j) -> v16h {
            const int dIdx = wd * 128 + j * 16 + nl;
            const uint32_t* rowp = ebuf + dIdx * LROW + (laneHi ? 8 : 0);
            union { uint4 q[2]; v16h v; } u;
            u.q[0] = ((const uint4*)rowp)[0];
            u.q[1] = ((const uint4*)rowp)[1];
            return u.v;
        };

        // prefetch first B fragment; its latency hides under the A build below
        v16h bmb[2];
        bmb[0] = loadBM(0);

        // ---- build 2 A fragments: w[tA, kbase .. kbase+31] in f16 ----
        // A k order here (contiguous, matches B): lanes<16: V0..7 -> K 0..15
        //                        lanes>=16: V0..7 -> K 16..31
        const int kbase = ic * 32;
        v16h a[2];
        #pragma unroll
        for (int ar = 0; ar < 2; ++ar) {
            const float tf = tfv[ar], in = inrm[ar];
            const bool fp = fpv[ar];
            #pragma unroll
            for (int v = 0; v < 8; ++v) {
                const int kp = v + (laneHi ? 8 : 0);
                const int n0 = kbase + 2 * kp;
                const int n1 = n0 + 1;
                float z0 = (tf - sc[n0]) * sv[n0];
                float z1 = (tf - sc[n1]) * sv[n1];
                float w0 = __expf(fmaf(-0.5f * z0, z0, sl[n0])) * in;
                float w1 = __expf(fmaf(-0.5f * z1, z1, sl[n1])) * in;
                if (fp) { w0 = (n0 == NTOK - 1) ? 1.f : 0.f;
                          w1 = (n1 == NTOK - 1) ? 1.f : 0.f; }
                a[ar][2 * v]     = (_Float16)w0;
                a[ar][2 * v + 1] = (_Float16)w1;
            }
        }

        // ---- software-pipelined B fragments + 16 WMMAs ----
        #pragma unroll
        for (int j = 0; j < 8; ++j) {
            if (j + 1 < 8) bmb[(j + 1) & 1] = loadBM(j + 1);
            acc[0][j] = __builtin_amdgcn_wmma_f32_16x16x32_f16(
                false, a[0], false, bmb[j & 1], (short)0, acc[0][j], false, false);
            acc[1][j] = __builtin_amdgcn_wmma_f32_16x16x32_f16(
                false, a[1], false, bmb[j & 1], (short)0, acc[1][j], false, false);
        }
    }

    // ---- store D: lanes<16: N=lane, M=r ; lanes>=16: N=lane-16, M=8+r ----
    const int nD = lane & 15;
    const int mBase = laneHi ? 8 : 0;
    #pragma unroll
    for (int ar = 0; ar < 2; ++ar) {
        #pragma unroll
        for (int j = 0; j < 8; ++j) {
            const int d = d0 + wd * 128 + j * 16 + nD;
            #pragma unroll
            for (int r = 0; r < 8; ++r) {
                const int t = t0 + wt * 32 + ar * 16 + mBase + r;
                out[((size_t)b * TT + t) * DD + d] = acc[ar][j][r];
            }
        }
    }
}

// ------------------------------ host launcher ------------------------------
extern "C" void kernel_launch(void* const* d_in, const int* in_sizes, int n_in,
                              void* d_out, int out_size, void* d_ws, size_t ws_size,
                              hipStream_t stream)
{
    const int*   text    = (const int*)  d_in[0];   // [B, 512]
    const int*   durs    = (const int*)  d_in[1];   // [B, 512]
    const float* embed_w = (const float*)d_in[2];   // [100, 512]
    (void)in_sizes; (void)n_in; (void)out_size; (void)ws_size;

    // workspace carve-up (all 256B aligned sizes)
    char* ws = (char*)d_ws;
    size_t off = 0;
    float* gc    = (float*)(ws + off); off += (size_t)BB * NP * 4;
    float* ginv  = (float*)(ws + off); off += (size_t)BB * NP * 4;
    float* glse  = (float*)(ws + off); off += (size_t)BB * NP * 4;
    float* gends = (float*)(ws + off); off += (size_t)BB * NP * 4;
    int*   gtok  = (int*)  (ws + off); off += (size_t)BB * NP * 4;
    float* ginvn = (float*)(ws + off); off += (size_t)BB * TT * 4;
    float* gfpad = (float*)(ws + off); off += (size_t)BB * TT * 4;
    uint32_t* ebufT = (uint32_t*)(ws + off);                          // ~17.8MB

    float* out = (float*)d_out;

    ge_params<<<BB, 256, 0, stream>>>(text, durs, gc, ginv, glse, gends, gtok);

    {
        const int total = BB * DD * EROW;
        ge_embT<<<(total + 255) / 256, 256, 0, stream>>>(embed_w, gtok, ebufT);
    }

    {
        dim3 grid(TT / 128, BB);
        ge_rows<<<grid, 128, 0, stream>>>(gc, ginv, glse, gends, gtok, ginvn, gfpad);
    }

    {
        dim3 grid(DD / 256, TT / 128, BB);
        ge_gemm<<<grid, 256, 0, stream>>>(gc, ginv, glse, ginvn, gfpad, ebufT, out);
    }
}
